// RNN_27693949124853
// MI455X (gfx1250) — matrix operand order repaired and checked by reading, not compiled, not run on hardware
//
#include <hip/hip_runtime.h>
#include <hip/hip_bf16.h>
#include <math.h>

// ---------------------------------------------------------------------------
// RNN forward on MI455X (gfx1250, wave32, WMMA).
//   Phase 0: pack W_x / W_h (f32 -> bf16) into WMMA B-fragment lane layout,
//            combine biases.
//   Phase 1: xw = x @ W_x + (b_x + b_h)   (65536x512x512 bf16 WMMA GEMM).
//            A tile staged block-wide into LDS via async-tensor DMA
//            (global_load_async_to_lds_b128 + s_wait_asynccnt), removing the
//            8x redundant per-wave A fetch. Output stored [T][B][H] f32.
//   Phase 2: sequential scan h = tanh(xw_t + h @ W_h). Batch-split: 4
//            independent workgroups (16 rows each), 32 waves, W_h resident
//            in VGPRs for all 1024 steps, h exchanged via double-buffered LDS.
//            xw C-addend loads software-pipelined one step ahead.
//   Phase 3: out = h_T @ W_fc + b_fc (tiny GEMV).
// ---------------------------------------------------------------------------

typedef __attribute__((ext_vector_type(16))) __bf16 v16bf;
typedef __attribute__((ext_vector_type(8)))  float  v8f;
typedef __attribute__((address_space(3))) char* lds_ptr_t;

#define HID   512
#define TLEN  1024
#define BATCH 64

// workspace layout (bytes); total ~135.5 MB
#define WS_WX 0                               // 512*512 bf16 packed  (512 KB)
#define WS_WH (512*1024)                      // 512*512 bf16 packed  (512 KB)
#define WS_BC (1024*1024)                     // 512 f32 combined bias
#define WS_HT (WS_BC + 8192)                  // 64*512 f32 h_T
#define WS_XW (WS_HT + BATCH*HID*4)           // 1024*64*512 f32 xw   (128 MB)

// --- Phase 0a: pack a 512x512 f32 matrix into bf16 B-fragment layout --------
// packed[((kt*32 + nt)*32 + lane)*16 + j] = W[kt*32 + (j<8 ? (lane>>4)*8 + j : 16 + (lane>>4)*8 + (j-8))][nt*16 + (lane&15)]
__global__ void pack_w_bf16(const float* __restrict__ src, __bf16* __restrict__ dst) {
    int idx  = blockIdx.x * blockDim.x + threadIdx.x;   // 0 .. 512*512-1
    int j    = idx & 15;
    int lane = (idx >> 4) & 31;
    int tile = idx >> 9;
    int nt   = tile & 31;
    int kt   = tile >> 5;
    int k = kt * 32 + (j < 8 ? (lane >> 4) * 8 + j : 16 + (lane >> 4) * 8 + (j - 8));
    int n = nt * 16 + (lane & 15);
    dst[idx] = (__bf16)src[k * HID + n];
}

// --- Phase 0b: combined bias ------------------------------------------------
__global__ void bias_combine(const float* __restrict__ bx, const float* __restrict__ bh,
                             float* __restrict__ bc) {
    int i = blockIdx.x * blockDim.x + threadIdx.x;
    if (i < HID) bc[i] = bx[i] + bh[i];
}

// --- Phase 1: xw GEMM -------------------------------------------------------
// grid = 4096 M-tiles, block = 256 (8 waves); wave w computes M-tile blockIdx
// x N-tiles [4w, 4w+4). A tile (16x512 f32, 32 KB) staged once per block into
// LDS by async-tensor DMA; K-loop reads A from LDS, B streams from L2.
__global__ __launch_bounds__(256) void gemm_xw(const float* __restrict__ x,
                                               const __bf16* __restrict__ wxp,
                                               const float* __restrict__ bc,
                                               float* __restrict__ xw) {
    __shared__ alignas(16) float ashm[16 * HID];        // 32 KB A tile

    const int tid  = threadIdx.x;
    const int lane = tid & 31;
    const int w    = tid >> 5;
    const int mt   = blockIdx.x;
    const int nt0  = w * 4;
    const int kh   = lane >> 4;
    const int nl   = lane & 15;

    // ---- async stage: 2048 x 16B chunks, 8 per thread ----------------------
    {
        const float* gA = x + (size_t)mt * 16 * HID;
#pragma unroll
        for (int i = 0; i < 8; ++i) {
            int idx = tid + i * 256;                        // 16-byte chunk id
            const float* gp = gA + idx * 4;
            lds_ptr_t lp = (lds_ptr_t)ashm + idx * 16;
            asm volatile("global_load_async_to_lds_b128 %0, %1, off"
                         :: "v"(lp), "v"(gp) : "memory");
        }
        asm volatile("s_wait_asynccnt 0x0" ::: "memory");
    }
    __syncthreads();

    const float* rowf = ashm + nl * HID;                    // A row in LDS

    v8f acc[4] = {};
    for (int kt = 0; kt < 16; ++kt) {
        // A fragment: 16x32 bf16, built from two contiguous 8-float LDS runs
        const float* p0 = rowf + kt * 32 + kh * 8;
        const float* p1 = rowf + kt * 32 + 16 + kh * 8;
        float4 a0 = *(const float4*)p0;
        float4 a1 = *(const float4*)(p0 + 4);
        float4 a2 = *(const float4*)p1;
        float4 a3 = *(const float4*)(p1 + 4);
        v16bf av;
        av[0]=(__bf16)a0.x; av[1]=(__bf16)a0.y; av[2]=(__bf16)a0.z; av[3]=(__bf16)a0.w;
        av[4]=(__bf16)a1.x; av[5]=(__bf16)a1.y; av[6]=(__bf16)a1.z; av[7]=(__bf16)a1.w;
        av[8]=(__bf16)a2.x; av[9]=(__bf16)a2.y; av[10]=(__bf16)a2.z; av[11]=(__bf16)a2.w;
        av[12]=(__bf16)a3.x; av[13]=(__bf16)a3.y; av[14]=(__bf16)a3.z; av[15]=(__bf16)a3.w;

        // keep next K-slab of B in flight
        if (kt + 1 < 16)
            __builtin_prefetch(wxp + (size_t)(((kt + 1) * 32 + nt0) * 32 + lane) * 16, 0, 0);

#pragma unroll
        for (int i = 0; i < 4; ++i) {
            v16bf bfrag = *(const v16bf*)(wxp +
                (size_t)(((kt * 32 + (nt0 + i)) * 32 + lane)) * 16);
            acc[i] = __builtin_amdgcn_wmma_f32_16x16x32_bf16(
                false, av, false, bfrag, (short)0, acc[i], false, false);
        }
    }

#pragma unroll
    for (int i = 0; i < 4; ++i) {
        int   col  = (nt0 + i) * 16 + nl;
        float bias = bc[col];
#pragma unroll
        for (int v = 0; v < 8; ++v) {
            int row = mt * 16 + v + 8 * kh;     // = b*1024 + t
            int b   = row >> 10;
            int t   = row & 1023;
            xw[((size_t)(t * BATCH + b)) * HID + col] = acc[i][v] + bias;
        }
    }
}

// --- Phase 2: sequential scan ----------------------------------------------
// grid = 4 (batch tiles of 16 rows), block = 1024 (32 waves, wave w = N-tile w).
// W_h fragments (16 x v16bf = 128 VGPRs) stay resident in registers; xw
// C-addend loads are software-pipelined one timestep ahead.
__global__ __launch_bounds__(1024) void rnn_scan(const float* __restrict__ xw,
                                                 const __bf16* __restrict__ whp,
                                                 float* __restrict__ hT) {
    __shared__ alignas(32) __bf16 hbuf[2][16][32][16];   // double-buffered A-frags (32 KB)

    const int lane = threadIdx.x & 31;
    const int w    = threadIdx.x >> 5;
    const int g    = blockIdx.x;
    const int kh   = lane >> 4;
    const int nl   = lane & 15;
    const int n    = w * 16 + nl;           // global h column owned by this lane

    // writer mapping: element (m, k=n) of next step's A fragment
    const int kt_w = n >> 5;
    const int kl   = n & 31;
    int khw, jw;
    if (kl < 16) { khw = kl >> 3;        jw = kl & 7; }
    else         { khw = (kl >> 3) & 1;  jw = 8 + (kl & 7); }

    // preload W_h B-fragments: resident in VGPRs for the whole scan
    v16bf breg[16];
#pragma unroll
    for (int kt = 0; kt < 16; ++kt)
        breg[kt] = *(const v16bf*)(whp + (size_t)(((kt * 32 + w) * 32 + lane)) * 16);

    // per-v base pointers into xw (row fixed, advance by one timestep = 32768 f32)
    const float* xwp[8];
#pragma unroll
    for (int v = 0; v < 8; ++v)
        xwp[v] = xw + (size_t)(g * 16 + v + 8 * kh) * HID + n;

    // h0 = 0
    {
        __bf16* hz = &hbuf[0][0][0][0];
        for (int i = threadIdx.x; i < 16 * 32 * 16; i += 1024) hz[i] = (__bf16)0.0f;
    }
    __syncthreads();

    // prologue: preload xw for t=0
    v8f cnext;
#pragma unroll
    for (int v = 0; v < 8; ++v) cnext[v] = xwp[v][0];

    int buf = 0;
    for (int t = 0; t < TLEN; ++t) {
        v8f acc = cnext;
        // issue next step's xw loads now; they complete under the WMMA chain
        if (t + 1 < TLEN) {
#pragma unroll
            for (int v = 0; v < 8; ++v)
                cnext[v] = xwp[v][(size_t)(t + 1) * BATCH * HID];
        }
        // h @ W_h : 16 WMMAs, A from LDS, B from resident VGPRs
#pragma unroll
        for (int kt = 0; kt < 16; ++kt) {
            v16bf afrag = *(const v16bf*)&hbuf[buf][kt][lane][0];
            acc = __builtin_amdgcn_wmma_f32_16x16x32_bf16(
                false, afrag, false, breg[kt], (short)0, acc, false, false);
        }
        // tanh, scatter into next step's A-fragment layout
        const int nbuf = buf ^ 1;
#pragma unroll
        for (int v = 0; v < 8; ++v) {
            float th = tanhf(acc[v]);
            int   m  = v + 8 * kh;
            hbuf[nbuf][kt_w][m + 16 * khw][jw] = (__bf16)th;
            if (t == TLEN - 1) hT[(size_t)(g * 16 + m) * HID + n] = th;
        }
        __syncthreads();
        buf = nbuf;
    }
}

// --- Phase 3: out = h_T @ W_fc + b_fc --------------------------------------
__global__ void fc_out(const float* __restrict__ hT, const float* __restrict__ wfc,
                       const float* __restrict__ bfc, float* __restrict__ out) {
    int b = threadIdx.x;
    if (b < BATCH) {
        float s = bfc[0];
        for (int k = 0; k < HID; ++k) s += hT[b * HID + k] * wfc[k];
        out[b] = s;
    }
}

extern "C" void kernel_launch(void* const* d_in, const int* in_sizes, int n_in,
                              void* d_out, int out_size, void* d_ws, size_t ws_size,
                              hipStream_t stream) {
    (void)in_sizes; (void)n_in; (void)out_size; (void)ws_size;
    const float* x    = (const float*)d_in[0];
    const float* W_x  = (const float*)d_in[1];
    const float* b_x  = (const float*)d_in[2];
    const float* W_h  = (const float*)d_in[3];
    const float* b_h  = (const float*)d_in[4];
    const float* W_fc = (const float*)d_in[5];
    const float* b_fc = (const float*)d_in[6];

    char*    ws  = (char*)d_ws;
    __bf16*  wxp = (__bf16*)(ws + WS_WX);
    __bf16*  whp = (__bf16*)(ws + WS_WH);
    float*   bc  = (float*)(ws + WS_BC);
    float*   hT  = (float*)(ws + WS_HT);
    float*   xw  = (float*)(ws + WS_XW);
    float*   out = (float*)d_out;

    pack_w_bf16 <<<(HID * HID) / 256, 256, 0, stream>>>(W_x, wxp);
    pack_w_bf16 <<<(HID * HID) / 256, 256, 0, stream>>>(W_h, whp);
    bias_combine<<<2, 256, 0, stream>>>(b_x, b_h, bc);
    gemm_xw     <<<(BATCH * TLEN) / 16, 256, 0, stream>>>(x, wxp, bc, xw);
    rnn_scan    <<<BATCH / 16, 1024, 0, stream>>>(xw, whp, hT);
    fc_out      <<<1, 64, 0, stream>>>(hT, W_fc, b_fc, out);
}
